// GAT_layer_v3_30459908063288
// MI455X (gfx1250) — matrix operand order repaired and checked by reading, not compiled, not run on hardware
//
#include <hip/hip_runtime.h>

// ---------------- problem constants ----------------
constexpr int kB = 2, kT = 12, kN = 512, kD = 256, kH = 8, kA = 32;
constexpr int kd = 32;               // head dim
constexpr int kSAMPLED = 108, kHALF = 54, kMED = 202, kSel = 216;
constexpr int kBH = kB * kH;         // 16
constexpr int kBT = kB * kT;         // 24
constexpr float kInvSqrtD = 0.17677669529663687f;  // 1/sqrt(32)

typedef __attribute__((ext_vector_type(16))) _Float16 v16h;
typedef __attribute__((ext_vector_type(8)))  float    v8f;

union HF { v16h v; _Float16 e[16]; uint4 q[2]; };
union AF { v8f  v; float    e[8]; };

__device__ inline v8f wmma16(v16h a, v16h b, v8f c) {
  // D = A(16x32 f16) * B(32x16 f16) + C(16x16 f32)
  return __builtin_amdgcn_wmma_f32_16x16x32_f16(false, a, false, b, (short)0, c, false, false);
}

// ---------------- WMMA GEMM: C[M,N] = alpha*A[M,K] @ B[N,K]^T + bias ----------------
// A,B f16 row-major. One wave -> one 16x16 tile; block = 8 waves -> 32x64 tile.
// Loads are branchless: row/col indices clamped into range (dup rows compute
// garbage that is never stored); only a single K-tail step carries guards.
__global__ void __launch_bounds__(256)
gemm_nt_kernel(const _Float16* __restrict__ A, const _Float16* __restrict__ Bm,
               const float* __restrict__ bias,
               float* __restrict__ C, _Float16* __restrict__ Ch,
               int M, int Nn, int K,
               long long sA, long long sB, long long sC,
               float alpha, int relu)
{
  int z = blockIdx.z;
  A  += (long long)z * sA;
  Bm += (long long)z * sB;
  long long cbase = (long long)z * sC;

  int wid  = threadIdx.x >> 5;
  int lane = threadIdx.x & 31;
  int half_ = lane >> 4;
  int l16   = lane & 15;
  int tileM = blockIdx.y * 32 + (wid >> 2) * 16;
  int tileN = blockIdx.x * 64 + (wid & 3) * 16;
  int rowA = min(tileM + l16, M - 1);     // clamped for loads only
  int colB = min(tileN + l16, Nn - 1);
  const _Float16* pA = A  + (long long)rowA * K + half_ * 8;
  const _Float16* pB = Bm + (long long)colB * K + half_ * 8;

  AF acc0, acc1;
#pragma unroll
  for (int r = 0; r < 8; ++r) { acc0.e[r] = 0.0f; acc1.e[r] = 0.0f; }

  const int Kmain = K & ~31;
  int kb = 0;
  // main loop: unroll x2, two accumulators -> back-to-back WMMA issue
  for (; kb + 64 <= Kmain; kb += 64) {
    HF fa0, fb0, fa1, fb1;
    fa0.q[0] = *(const uint4*)(pA + kb);
    fa0.q[1] = *(const uint4*)(pA + kb + 16);
    fb0.q[0] = *(const uint4*)(pB + kb);
    fb0.q[1] = *(const uint4*)(pB + kb + 16);
    fa1.q[0] = *(const uint4*)(pA + kb + 32);
    fa1.q[1] = *(const uint4*)(pA + kb + 48);
    fb1.q[0] = *(const uint4*)(pB + kb + 32);
    fb1.q[1] = *(const uint4*)(pB + kb + 48);
    if (kb + 192 <= Kmain) {              // prefetch 2 iterations ahead
      __builtin_prefetch(pA + kb + 128, 0, 1);
      __builtin_prefetch(pB + kb + 128, 0, 1);
    }
    acc0.v = wmma16(fa0.v, fb0.v, acc0.v);
    acc1.v = wmma16(fa1.v, fb1.v, acc1.v);
  }
  for (; kb < Kmain; kb += 32) {
    HF fa, fb;
    fa.q[0] = *(const uint4*)(pA + kb);
    fa.q[1] = *(const uint4*)(pA + kb + 16);
    fb.q[0] = *(const uint4*)(pB + kb);
    fb.q[1] = *(const uint4*)(pB + kb + 16);
    acc0.v = wmma16(fa.v, fb.v, acc0.v);
  }
  if (kb < K) {                           // K % 32 tail (K=216 case), runs once
    HF fa, fb;
#pragma unroll
    for (int e = 0; e < 8; ++e) {
      fa.e[e]     = (kb + half_ * 8 + e      < K) ? pA[kb + e]      : (_Float16)0.0f;
      fa.e[e + 8] = (kb + 16 + half_ * 8 + e < K) ? pA[kb + 16 + e] : (_Float16)0.0f;
    }
#pragma unroll
    for (int e = 0; e < 8; ++e) {
      fb.e[e]     = (kb + half_ * 8 + e      < K) ? pB[kb + e]      : (_Float16)0.0f;
      fb.e[e + 8] = (kb + 16 + half_ * 8 + e < K) ? pB[kb + 16 + e] : (_Float16)0.0f;
    }
    acc0.v = wmma16(fa.v, fb.v, acc0.v);
  }

  int col = tileN + l16;
  if (col < Nn) {
    float bb = bias ? bias[col] : 0.0f;
#pragma unroll
    for (int r = 0; r < 8; ++r) {
      int row = tileM + r + half_ * 8;
      if (row < M) {
        float v = alpha * (acc0.e[r] + acc1.e[r]) + bb;
        if (relu) v = fmaxf(v, 0.0f);
        long long off = cbase + (long long)row * Nn + col;
        if (C)  C[off]  = v;
        if (Ch) Ch[off] = (_Float16)v;
      }
    }
  }
}

// ---------------- reductions ----------------
__device__ inline float blockMaxF(float v, float* sred) {
  for (int o = 16; o > 0; o >>= 1) v = fmaxf(v, __shfl_xor(v, o, 32));
  int lane = threadIdx.x & 31, wid = threadIdx.x >> 5;
  if (lane == 0) sred[wid] = v;
  __syncthreads();
  int nw = ((int)blockDim.x + 31) >> 5;
  float r = (lane < nw) ? sred[lane] : -3.4e38f;
  for (int o = 16; o > 0; o >>= 1) r = fmaxf(r, __shfl_xor(r, o, 32));
  __syncthreads();
  return r;
}
__device__ inline float blockSumF(float v, float* sred) {
  for (int o = 16; o > 0; o >>= 1) v += __shfl_xor(v, o, 32);
  int lane = threadIdx.x & 31, wid = threadIdx.x >> 5;
  if (lane == 0) sred[wid] = v;
  __syncthreads();
  int nw = ((int)blockDim.x + 31) >> 5;
  float r = (lane < nw) ? sred[lane] : 0.0f;
  for (int o = 16; o > 0; o >>= 1) r += __shfl_xor(r, o, 32);
  __syncthreads();
  return r;
}

// ---------------- elementwise / relayout ----------------
__global__ void f32_to_f16_kernel(const float* __restrict__ src, _Float16* __restrict__ dst, long long n) {
  long long i = (long long)blockIdx.x * blockDim.x + threadIdx.x;
  if (i < n) dst[i] = (_Float16)src[i];
}

// src f16 [B,T,S,D] -> dst f16 [BH,T,S,32]   (bh = h*B + b)
__global__ void split_heads_kernel(const _Float16* __restrict__ src, _Float16* __restrict__ dst,
                                   int S, long long total) {
  long long i = (long long)blockIdx.x * blockDim.x + threadIdx.x;
  if (i >= total) return;
  int j = (int)(i & 31);
  long long r = i >> 5;
  int s = (int)(r % S);
  long long bt = r / S;
  int t = (int)(bt % kT);
  int bh = (int)(bt / kT);
  int h = bh >> 1, b = bh & 1;
  long long soff = (((long long)(b * kT + t) * S + s) * kD) + h * 32 + j;
  dst[i] = src[soff];
}

// src f16 [B,T,S,D] -> dst f16 [BH,T,32,S]   (head-split + transpose)
__global__ void split_heads_T_kernel(const _Float16* __restrict__ src, _Float16* __restrict__ dst,
                                     int S, long long total) {
  long long i = (long long)blockIdx.x * blockDim.x + threadIdx.x;
  if (i >= total) return;
  int s = (int)(i % S);
  long long r = i / S;
  int j = (int)(r & 31);
  long long bt = r >> 5;
  int t = (int)(bt % kT);
  int bh = (int)(bt / kT);
  int h = bh >> 1, b = bh & 1;
  long long soff = (((long long)(b * kT + t) * S + s) * kD) + h * 32 + j;
  dst[i] = src[soff];
}

// src f32 [BH,T,S,32] -> dstF f32 [B,T,S,D] and/or dstH f16
__global__ void merge_heads_kernel(const float* __restrict__ src, float* __restrict__ dstF,
                                   _Float16* __restrict__ dstH, int S, long long total) {
  long long i = (long long)blockIdx.x * blockDim.x + threadIdx.x;
  if (i >= total) return;
  int j = (int)(i & 31);
  long long r = i >> 5;
  int s = (int)(r % S);
  long long bt = r / S;
  int t = (int)(bt % kT);
  int bh = (int)(bt / kT);
  int h = bh >> 1, b = bh & 1;
  long long doff = (((long long)(b * kT + t) * S + s) * kD) + h * 32 + j;
  float v = src[i];
  if (dstF) dstF[doff] = v;
  if (dstH) dstH[doff] = (_Float16)v;
}

// M[bh,t,n] = sum_a (Q[bh,t,n,:] . K[bh,t,statica[n,a],:]) * pw[a] + pb
__global__ void sample_score_kernel(const _Float16* __restrict__ Qhm, const _Float16* __restrict__ Khm,
                                    const int* __restrict__ statica,
                                    const float* __restrict__ pw, const float* __restrict__ pb,
                                    float* __restrict__ Mb) {
  long long i = (long long)blockIdx.x * blockDim.x + threadIdx.x;
  if (i >= (long long)kBH * kT * kN) return;
  int n = (int)(i % kN);
  long long bt = i / kN;
  const _Float16* q = Qhm + i * kd;
  const _Float16* Kbase = Khm + bt * kN * kd;
  float acc = pb[0];
  for (int a = 0; a < kA; ++a) {
    int m = statica[n * kA + a];
    const _Float16* kk = Kbase + (long long)m * kd;
    float s = 0.0f;
#pragma unroll
    for (int j = 0; j < kd; ++j) s += (float)q[j] * (float)kk[j];
    acc += s * pw[a];
  }
  Mb[i] = acc;
}

// ---------------- argsort(512) + gumbel top-k sampling ----------------
__device__ inline unsigned pcg_hash(unsigned x) {
  x ^= x >> 17; x *= 0xED5AD4BBu; x ^= x >> 11; x *= 0xAC4C1B51u;
  x ^= x >> 15; x *= 0x31848BABu; x ^= x >> 14; return x;
}

__device__ void topk_region(const float* skey, const int* sidx, float* w2, int* i2,
                            int base, int cnt, unsigned row, unsigned salt,
                            int outK, int* outIdx, int tid) {
  if (tid < 256) {
    if (tid < cnt) {
      float v = skey[base + tid];
      if (v < 0.0f) v = 0.0f;                      // clip(., 0, None); ranking = clip(v)+gumbel
      unsigned h = pcg_hash(row * 0x9E3779B9u ^ salt * 0x85EBCA6Bu ^ (unsigned)tid * 0xC2B2AE35u ^ 42u);
      float u = ((float)h + 0.5f) * 2.3283064e-10f;
      float g = -logf(-logf(u));
      w2[tid] = -(v + g);                          // ascending sort -> largest first
      i2[tid] = sidx[base + tid];
    } else { w2[tid] = 3.0e38f; i2[tid] = 0x7FFFFFFF; }
  }
  __syncthreads();
  for (int k = 2; k <= 256; k <<= 1)
    for (int j = k >> 1; j > 0; j >>= 1) {
      if (tid < 256) {
        int ixj = tid ^ j;
        if (ixj > tid) {
          float a = w2[tid], b2 = w2[ixj];
          int ia = i2[tid], ib = i2[ixj];
          bool agtb = (a > b2) || (a == b2 && ia > ib);
          bool up = ((tid & k) == 0);
          bool sw = up ? agtb : ((b2 > a) || (b2 == a && ib > ia));
          if (sw) { w2[tid] = b2; w2[ixj] = a; i2[tid] = ib; i2[ixj] = ia; }
        }
      }
      __syncthreads();
    }
  if (tid < outK) outIdx[tid] = i2[tid];
  __syncthreads();
}

__global__ void __launch_bounds__(512)
sort_sample_kernel(const float* __restrict__ Mb, int* __restrict__ Msample) {
  __shared__ float skey[512];
  __shared__ int   sidx[512];
  __shared__ float w2[256];
  __shared__ int   i2[256];
  int row = blockIdx.x;                 // bh*T + t
  int tid = threadIdx.x;
  skey[tid] = Mb[(long long)row * kN + tid];
  sidx[tid] = tid;
  __syncthreads();
  for (int k = 2; k <= 512; k <<= 1)
    for (int j = k >> 1; j > 0; j >>= 1) {
      int ixj = tid ^ j;
      if (ixj > tid) {
        float a = skey[tid], b2 = skey[ixj];
        int ia = sidx[tid], ib = sidx[ixj];
        bool agtb = (a > b2) || (a == b2 && ia > ib);
        bool up = ((tid & k) == 0);
        bool sw = up ? agtb : ((b2 > a) || (b2 == a && ib > ia));
        if (sw) { skey[tid] = b2; skey[ixj] = a; sidx[tid] = ib; sidx[ixj] = ia; }
      }
      __syncthreads();
    }
  int* outRow = Msample + (long long)row * kSel;
  if (tid < kSAMPLED) outRow[tid] = sidx[511 - tid];      // top_k descending
  __syncthreads();
  topk_region(skey, sidx, w2, i2, kMED, kMED, (unsigned)row, 1u, kHALF, outRow + kSAMPLED, tid);
  topk_region(skey, sidx, w2, i2, 0,    kMED, (unsigned)row, 2u, kHALF, outRow + kSAMPLED + kHALF, tid);
}

// gather Q_red (head-major), projector (merged), K_red (merged)
__global__ void gather_kernel(const _Float16* __restrict__ Qf_h, const _Float16* __restrict__ Kf_h,
                              const int* __restrict__ Msample,
                              _Float16* __restrict__ QredA, _Float16* __restrict__ proj_h,
                              _Float16* __restrict__ Kred_h) {
  long long i = (long long)blockIdx.x * blockDim.x + threadIdx.x;
  if (i >= (long long)kBH * kT * kSel * kd) return;
  int j = (int)(i & 31);
  long long r = i >> 5;                 // bh*T*216 + t*216 + s
  int s = (int)(r % kSel);
  long long bt = r / kSel;
  int t = (int)(bt % kT);
  int bh = (int)(bt / kT);
  int h = bh >> 1, b = bh & 1;
  int n = Msample[r];
  long long srcOff = (((long long)(b * kT + t) * kN + n) * kD) + h * 32 + j;
  long long mOff   = (((long long)(b * kT + t) * kSel + s) * kD) + h * 32 + j;
  _Float16 qv = Qf_h[srcOff];
  QredA[i] = qv;
  proj_h[mOff] = qv;
  Kred_h[mOff] = Kf_h[srcOff];
}

// row softmax over last dim L (in-place f32, optional f16 copy)
__global__ void softmax_kernel(float* __restrict__ X, _Float16* __restrict__ Xh, int L) {
  __shared__ float sred[33];
  long long row = blockIdx.x;
  float* x = X + row * L;
  int tid = threadIdx.x;
  float m = -3.4e38f;
  for (int i = tid; i < L; i += blockDim.x) m = fmaxf(m, x[i]);
  m = blockMaxF(m, sred);
  float s = 0.0f;
  for (int i = tid; i < L; i += blockDim.x) s += expf(x[i] - m);
  s = blockSumF(s, sred);
  float inv = 1.0f / s;
  for (int i = tid; i < L; i += blockDim.x) {
    float v = expf(x[i] - m) * inv;
    x[i] = v;
    if (Xh) Xh[row * L + i] = (_Float16)v;
  }
}

// layernorm over D=256 (one row per block of 256 threads), optional residual/affine
__global__ void __launch_bounds__(256)
layernorm_kernel(const float* __restrict__ X, const float* __restrict__ R,
                 const float* __restrict__ gw, const float* __restrict__ gb,
                 float* __restrict__ outF, _Float16* __restrict__ outH) {
  __shared__ float sred[33];
  long long row = blockIdx.x;
  int tid = threadIdx.x;
  float v = X[row * kD + tid];
  if (R) v += R[row * kD + tid];
  float mu = blockSumF(v, sred) * (1.0f / kD);
  float dv = v - mu;
  float var = blockSumF(dv * dv, sred) * (1.0f / kD);
  float y = dv * rsqrtf(var + 1e-5f);
  if (gw) y = y * gw[tid] + gb[tid];
  if (outF) outF[row * kD + tid] = y;
  if (outH) outH[row * kD + tid] = (_Float16)y;
}

// ---------------- host driver ----------------
static inline int ceilDiv(int a, int b) { return (a + b - 1) / b; }

extern "C" void kernel_launch(void* const* d_in, const int* in_sizes, int n_in,
                              void* d_out, int out_size, void* d_ws, size_t ws_size,
                              hipStream_t stream) {
  (void)in_sizes; (void)n_in; (void)out_size; (void)ws_size;
  const float* x    = (const float*)d_in[0];
  const int*   stat = (const int*)d_in[1];
  const float* Wq = (const float*)d_in[2],  *bq = (const float*)d_in[3];
  const float* Wk = (const float*)d_in[4],  *bk = (const float*)d_in[5];
  const float* Wv = (const float*)d_in[6],  *bv = (const float*)d_in[7];
  const float* Wsk = (const float*)d_in[8], *bsk = (const float*)d_in[9];
  const float* Wsv = (const float*)d_in[10],*bsv = (const float*)d_in[11];
  const float* Wo = (const float*)d_in[12], *bo = (const float*)d_in[13];
  const float* Wf1 = (const float*)d_in[14],*bf1 = (const float*)d_in[15];
  const float* Wf2 = (const float*)d_in[16],*bf2 = (const float*)d_in[17];
  const float* pw = (const float*)d_in[18], *pb = (const float*)d_in[19];
  const float* lnw = (const float*)d_in[20],*lnb = (const float*)d_in[21];

  float* OUT_out  = (float*)d_out;                                   // [B,T,N,D]
  float* OUT_Apro = OUT_out  + (long long)kB * kT * kN * kD;         // [B,T,N,216]
  float* OUT_Mpro = OUT_Apro + (long long)kB * kT * kN * kSel;       // [B,T,216,D]

  char* wp = (char*)d_ws;
  auto take = [&](long long bytes) -> char* {
    char* p = wp; wp += (bytes + 255) & ~255LL; return p;
  };
  const long long nTok = (long long)kB * kT * kN;          // 12288
  const long long nXD  = nTok * kD;                        // 3,145,728
  const long long nSelD = (long long)kBT * kSel * kD;      // 1,327,104
  const long long nHm  = (long long)kBH * kT * kN * kd;    // 3,145,728
  const long long nRedHm = (long long)kBH * kT * kSel * kd;// 1,327,104

  _Float16* xh   = (_Float16*)take(nXD * 2);
  _Float16* Wh   = (_Float16*)take(8LL * kD * kD * 2);
  _Float16* Qf_h = (_Float16*)take(nXD * 2);
  _Float16* Kf_h = (_Float16*)take(nXD * 2);
  _Float16* Vf_h = (_Float16*)take(nXD * 2);
  _Float16* Q_hm = (_Float16*)take(nHm * 2);
  _Float16* K_hm = (_Float16*)take(nHm * 2);
  _Float16* Vt_hm = (_Float16*)take(nHm * 2);              // [BH,T,32,N]
  float*    Mbuf = (float*)take((long long)kBH * kT * kN * 4);
  int*      Msam = (int*)take((long long)kBH * kT * kSel * 4);
  _Float16* QredA = (_Float16*)take(nRedHm * 2);
  _Float16* proj_h = (_Float16*)take(nSelD * 2);
  _Float16* Kred_h = (_Float16*)take(nSelD * 2);
  float*    scoresF = (float*)take((long long)kT * kSel * kN * 4);   // reused scratch
  _Float16* scoresH = (_Float16*)take((long long)kT * kSel * kN * 2);
  float*    val_hm = (float*)take(nRedHm * 4);
  _Float16* valm_h = (_Float16*)take(nSelD * 2);
  _Float16* sK_h = (_Float16*)take(nSelD * 2);
  _Float16* sV_h = (_Float16*)take(nSelD * 2);
  _Float16* sK_hm = (_Float16*)take(nRedHm * 2);
  _Float16* sVt_hm = (_Float16*)take(nRedHm * 2);          // [BH,T,32,216]
  float*    v2_hm = (float*)take(nHm * 4);
  _Float16* v2m_h = (_Float16*)take(nXD * 2);
  float*    v2o   = (float*)take(nXD * 4);
  float*    v2ln  = (float*)take(nXD * 4);
  _Float16* v2ln_h = (_Float16*)take(nXD * 2);
  _Float16* hid_h = (_Float16*)take(nXD * 2);
  float*    hf    = (float*)take(nXD * 4);

  dim3 blk(256);
  auto ew = [&](long long n) { return dim3((unsigned)((n + 255) / 256)); };
  auto gg = [&](int M, int Nn, int batch) { return dim3(ceilDiv(Nn, 64), ceilDiv(M, 32), batch); };

  // 1) f32 -> f16
  f32_to_f16_kernel<<<ew(nXD), blk, 0, stream>>>(x, xh, nXD);
  const float* Wsrc[8] = {Wq, Wk, Wv, Wsk, Wsv, Wo, Wf1, Wf2};
  for (int i = 0; i < 8; ++i)
    f32_to_f16_kernel<<<ew(kD * kD), blk, 0, stream>>>(Wsrc[i], Wh + (long long)i * kD * kD, kD * kD);
  _Float16 *WqH = Wh, *WkH = Wh + 65536, *WvH = Wh + 2 * 65536, *WskH = Wh + 3 * 65536,
           *WsvH = Wh + 4 * 65536, *WoH = Wh + 5 * 65536, *Wf1H = Wh + 6 * 65536, *Wf2H = Wh + 7 * 65536;

  // 2) Q/K/V projections (WMMA): [12288,256] = xh @ W^T + b
  gemm_nt_kernel<<<gg(nTok, kD, 1), blk, 0, stream>>>(xh, WqH, bq, nullptr, Qf_h, nTok, kD, kD, 0, 0, 0, 1.0f, 0);
  gemm_nt_kernel<<<gg(nTok, kD, 1), blk, 0, stream>>>(xh, WkH, bk, nullptr, Kf_h, nTok, kD, kD, 0, 0, 0, 1.0f, 0);
  gemm_nt_kernel<<<gg(nTok, kD, 1), blk, 0, stream>>>(xh, WvH, bv, nullptr, Vf_h, nTok, kD, kD, 0, 0, 0, 1.0f, 0);

  // 3) split heads -> [BH,T,N,32] (Q,K) and transposed [BH,T,32,N] (V)
  split_heads_kernel<<<ew(nHm), blk, 0, stream>>>(Qf_h, Q_hm, kN, nHm);
  split_heads_kernel<<<ew(nHm), blk, 0, stream>>>(Kf_h, K_hm, kN, nHm);
  split_heads_T_kernel<<<ew(nHm), blk, 0, stream>>>(Vf_h, Vt_hm, kN, nHm);

  // 4) sampled scores M, 5) sort + sample -> Msample
  sample_score_kernel<<<ew((long long)kBH * kT * kN), blk, 0, stream>>>(Q_hm, K_hm, stat, pw, pb, Mbuf);
  sort_sample_kernel<<<dim3(kBH * kT), dim3(512), 0, stream>>>(Mbuf, Msam);

  // 6) gathers
  gather_kernel<<<ew(nRedHm), blk, 0, stream>>>(Qf_h, Kf_h, Msam, QredA, proj_h, Kred_h);

  // 7) A_pro = softmax(Qf @ K_red^T / sqrt(d))  -> directly into d_out
  gemm_nt_kernel<<<gg(kN, kSel, kBT), blk, 0, stream>>>(
      Qf_h, Kred_h, nullptr, OUT_Apro, nullptr, kN, kSel, kD,
      (long long)kN * kD, (long long)kSel * kD, (long long)kN * kSel, kInvSqrtD, 0);
  softmax_kernel<<<dim3((unsigned)(kBT * kN)), blk, 0, stream>>>(OUT_Apro, nullptr, kSel);

  // 8) attention: per head-batch bh, batched over T (all WMMA NT GEMMs)
  for (int bh = 0; bh < kBH; ++bh) {
    long long hmOff  = (long long)bh * kT * kN * kd;
    long long redOff = (long long)bh * kT * kSel * kd;
    gemm_nt_kernel<<<gg(kSel, kN, kT), blk, 0, stream>>>(
        QredA + redOff, K_hm + hmOff, nullptr, scoresF, nullptr, kSel, kN, kd,
        (long long)kSel * kd, (long long)kN * kd, (long long)kSel * kN, kInvSqrtD, 0);
    softmax_kernel<<<dim3((unsigned)(kT * kSel)), blk, 0, stream>>>(scoresF, scoresH, kN);
    // val[216,32] = attn[216,512] @ V[512,32] == NT with Vt rows as B
    gemm_nt_kernel<<<gg(kSel, kd, kT), blk, 0, stream>>>(
        scoresH, Vt_hm + hmOff, nullptr, val_hm + redOff, nullptr, kSel, kd, kN,
        (long long)kSel * kN, (long long)kd * kN, (long long)kSel * kd, 1.0f, 0);
  }

  // 9) merge val -> M_pro (f32 out) + valm_h (f16)
  merge_heads_kernel<<<ew(nRedHm), blk, 0, stream>>>(val_hm, OUT_Mpro, valm_h, kSel, nRedHm);

  // 10) sK / sV projections
  gemm_nt_kernel<<<gg((int)(kBT * kSel), kD, 1), blk, 0, stream>>>(
      proj_h, WskH, bsk, nullptr, sK_h, kBT * kSel, kD, kD, 0, 0, 0, 1.0f, 0);
  gemm_nt_kernel<<<gg((int)(kBT * kSel), kD, 1), blk, 0, stream>>>(
      valm_h, WsvH, bsv, nullptr, sV_h, kBT * kSel, kD, kD, 0, 0, 0, 1.0f, 0);
  split_heads_kernel<<<ew(nRedHm), blk, 0, stream>>>(sK_h, sK_hm, kSel, nRedHm);
  split_heads_T_kernel<<<ew(nRedHm), blk, 0, stream>>>(sV_h, sVt_hm, kSel, nRedHm);

  // 11) sQK = softmax(Q @ sK^T / sqrt(d)); v2 = sQK @ sV
  for (int bh = 0; bh < kBH; ++bh) {
    long long hmOff  = (long long)bh * kT * kN * kd;
    long long redOff = (long long)bh * kT * kSel * kd;
    gemm_nt_kernel<<<gg(kN, kSel, kT), blk, 0, stream>>>(
        Q_hm + hmOff, sK_hm + redOff, nullptr, scoresF, nullptr, kN, kSel, kd,
        (long long)kN * kd, (long long)kSel * kd, (long long)kN * kSel, kInvSqrtD, 0);
    softmax_kernel<<<dim3((unsigned)(kT * kN)), blk, 0, stream>>>(scoresF, scoresH, kSel);
    // v2[512,32] = sQK[512,216] @ sV[216,32] == NT with sVt rows as B (K=216 tail path)
    gemm_nt_kernel<<<gg(kN, kd, kT), blk, 0, stream>>>(
        scoresH, sVt_hm + redOff, nullptr, v2_hm + hmOff, nullptr, kN, kd, kSel,
        (long long)kN * kSel, (long long)kd * kSel, (long long)kN * kd, 1.0f, 0);
  }
  merge_heads_kernel<<<ew(nHm), blk, 0, stream>>>(v2_hm, nullptr, v2m_h, kN, nHm);

  // 12) v2 = v2m @ Wo^T + bo ; ln affine ; FFN ; final residual LN
  gemm_nt_kernel<<<gg((int)nTok, kD, 1), blk, 0, stream>>>(
      v2m_h, WoH, bo, v2o, nullptr, (int)nTok, kD, kD, 0, 0, 0, 1.0f, 0);
  layernorm_kernel<<<dim3((unsigned)nTok), blk, 0, stream>>>(v2o, nullptr, lnw, lnb, v2ln, v2ln_h);
  gemm_nt_kernel<<<gg((int)nTok, kD, 1), blk, 0, stream>>>(
      v2ln_h, Wf1H, bf1, nullptr, hid_h, (int)nTok, kD, kD, 0, 0, 0, 1.0f, 1);
  gemm_nt_kernel<<<gg((int)nTok, kD, 1), blk, 0, stream>>>(
      hid_h, Wf2H, bf2, hf, nullptr, (int)nTok, kD, kD, 0, 0, 0, 1.0f, 0);
  layernorm_kernel<<<dim3((unsigned)nTok), blk, 0, stream>>>(hf, v2ln, nullptr, nullptr, OUT_out, nullptr);
}
